// GNN2_2946347565063
// MI455X (gfx1250) — matrix operand order repaired and checked by reading, not compiled, not run on hardware
//
#include <hip/hip_runtime.h>
#include <hip/hip_bf16.h>

typedef __attribute__((ext_vector_type(16))) _Float16 v16h;
typedef __attribute__((ext_vector_type(8)))  _Float16 v8h;
typedef __attribute__((ext_vector_type(8)))  float    v8f;

#define NEG_ATT 0.2f
#define NEG_ACT 0.02f

// ---------------------------------------------------------------------------
// Monotone float <-> uint key mapping so atomicMax(uint) orders like floats.
// key(f) is strictly increasing in f; key==0 is below every finite float,
// and every dst node has at least its self-loop edge, so 0-init is safe.
// ---------------------------------------------------------------------------
__device__ __forceinline__ unsigned fkey(float f) {
    unsigned b = __float_as_uint(f);
    return (b & 0x80000000u) ? ~b : (b | 0x80000000u);
}
__device__ __forceinline__ float fdec(unsigned k) {
    unsigned b = (k & 0x80000000u) ? (k ^ 0x80000000u) : ~k;
    return __uint_as_float(b);
}

// ---------------------------------------------------------------------------
// Small dense GEMM: O[N,M] = A[N,K] @ W[K,M]   (f32, one thread per output)
// ---------------------------------------------------------------------------
__global__ void gemm_f32(const float* __restrict__ A, const float* __restrict__ W,
                         float* __restrict__ O, int N, int K, int M) {
    int idx = blockIdx.x * blockDim.x + threadIdx.x;
    if (idx >= N * M) return;
    int i = idx / M, j = idx % M;
    const float* a = A + (size_t)i * K;
    float s = 0.f;
    for (int k = 0; k < K; ++k) s = fmaf(a[k], W[(size_t)k * M + j], s);
    O[idx] = s;
}

// alpha_src[i] = h[i]·a_src ; alpha_dst[i] = h[i]·a_dst (one thread per node)
__global__ void node_alphas(const float* __restrict__ H, const float* __restrict__ as,
                            const float* __restrict__ ad, float* __restrict__ As,
                            float* __restrict__ Ad, int N, int M) {
    int i = blockIdx.x * blockDim.x + threadIdx.x;
    if (i >= N) return;
    const float* h = H + (size_t)i * M;
    float s = 0.f, d = 0.f;
    for (int k = 0; k < M; ++k) { s = fmaf(h[k], as[k], s); d = fmaf(h[k], ad[k], d); }
    As[i] = s; Ad[i] = d;
}

// Zero agg[N*M] (f32), denom[N] (f32), mkey[N] (uint)
__global__ void init_layer(float* __restrict__ agg, float* __restrict__ denom,
                           unsigned* __restrict__ mkey, int N, int M) {
    int idx = blockIdx.x * blockDim.x + threadIdx.x;
    int total = N * M;
    if (idx < total) agg[idx] = 0.f;
    if (idx < N) { denom[idx] = 0.f; mkey[idx] = 0u; }
}

// Pass 1: e = leaky_relu(As[src]+Ad[dst], 0.2); segment-max via uint atomicMax.
// Edges [0,E) come from edge_index; [E, E+N) are the appended self loops.
__global__ void edge_logits_max(const int* __restrict__ src, const int* __restrict__ dst,
                                const float* __restrict__ As, const float* __restrict__ Ad,
                                float* __restrict__ eV, unsigned* __restrict__ mkey,
                                int E, int Etot) {
    int e = blockIdx.x * blockDim.x + threadIdx.x;
    if (e >= Etot) return;
    int s, d;
    if (e < E) { s = src[e]; d = dst[e]; } else { s = d = e - E; }
    float v = As[s] + Ad[d];
    v = (v > 0.f) ? v : NEG_ATT * v;
    eV[e] = v;
    atomicMax(mkey + d, fkey(v));
}

// Pass 2: ex = exp(e - m[dst]); segment-sum via f32 atomicAdd.
__global__ void edge_expsum(const int* __restrict__ dst, const float* __restrict__ eV,
                            const unsigned* __restrict__ mkey, float* __restrict__ exV,
                            float* __restrict__ denom, int E, int Etot) {
    int e = blockIdx.x * blockDim.x + threadIdx.x;
    if (e >= Etot) return;
    int d = (e < E) ? dst[e] : (e - E);
    float x = __expf(eV[e] - fdec(mkey[d]));
    exV[e] = x;
    atomicAdd(denom + d, x);
}

// Pass 3: agg[dst,c] += (ex/denom[dst]) * h[src,c]  (one thread per edge-channel)
__global__ void edge_aggregate(const int* __restrict__ src, const int* __restrict__ dst,
                               const float* __restrict__ exV, const float* __restrict__ denom,
                               const float* __restrict__ H, float* __restrict__ agg,
                               int E, int Etot, int M) {
    long long t = (long long)blockIdx.x * blockDim.x + threadIdx.x;
    long long total = (long long)Etot * M;
    if (t >= total) return;
    int e = (int)(t / M), c = (int)(t % M);
    int s, d;
    if (e < E) { s = src[e]; d = dst[e]; } else { s = d = e - E; }
    float alpha = exV[e] / denom[d];
    atomicAdd(agg + (size_t)d * M + c, alpha * H[(size_t)s * M + c]);
}

// out = leaky_relu(agg + b, 0.02); optionally also emit f16 copy for WMMA GEMM.
__global__ void bias_act(const float* __restrict__ agg, const float* __restrict__ b,
                         float* __restrict__ out32, _Float16* __restrict__ out16,
                         int N, int M) {
    int idx = blockIdx.x * blockDim.x + threadIdx.x;
    if (idx >= N * M) return;
    float v = agg[idx] + b[idx % M];
    v = (v > 0.f) ? v : NEG_ACT * v;
    if (out32) out32[idx] = v;
    if (out16) out16[idx] = (_Float16)v;
}

// ---------------------------------------------------------------------------
// C[N,N] = H @ H^T, H f16 [N,128], f32 accumulate via v_wmma_f32_16x16x32_f16.
//
// LDS-blocked: each 256-thread block (8 waves) computes a 128x128 C tile.
// The two 128x128(K) f16 strips of H are staged once into LDS (32KB+32KB),
// so global reads = 4096 blocks * 64KB = 268MB, balanced with the 268MB of
// NT output writes. Per wave: A rows held in registers (4x v16h), 8 N-tiles
// x 4 K-chunks = 32 WMMAs, 8x v8f accumulators.
//
// WMMA operand layouts used here (contiguous k, not the ISA table order; A and B agree):
//  A 16x32 f16 : lane<16 -> M=lane,   K0..15 contiguous
//                lane>=16 -> M=lane-16, K16..31 contiguous
//  B 32x16 f16 : lane<16 -> N=lane,    K0..15 contiguous
//                lane>=16 -> N=lane-16, K16..31 contiguous  (B[k,n]=H[n,k])
//  C/D 16x16 f32: VGPR r: lanes 0-15 -> (M=r, N=lane); 16-31 -> (M=r+8, N=l)
// ---------------------------------------------------------------------------
#define HHK 128
__global__ void __launch_bounds__(256) hhT_wmma(const _Float16* __restrict__ Hf,
                                                float* __restrict__ C, int N) {
    __shared__ _Float16 Ash[128 * HHK];   // 32 KB
    __shared__ _Float16 Bsh[128 * HHK];   // 32 KB

    const int m_blk = blockIdx.x * 128;
    const int n_blk = blockIdx.y * 128;

    // Cooperative stage: 128 rows * 128 halves; each thread copies 8 halves
    // (16B) per iteration, 8 iterations per buffer.
#pragma unroll
    for (int i = threadIdx.x; i < (128 * HHK) / 8; i += 256) {
        int elem = i * 8;
        int r = elem >> 7, c = elem & (HHK - 1);
        *(v8h*)&Ash[elem] = *(const v8h*)&Hf[(size_t)(m_blk + r) * HHK + c];
        *(v8h*)&Bsh[elem] = *(const v8h*)&Hf[(size_t)(n_blk + r) * HHK + c];
    }
    __syncthreads();

    const int lane = threadIdx.x & 31;
    const int wave = threadIdx.x >> 5;
    const int half = lane >> 4;          // 0|1
    const int l    = lane & 15;

    // Preload this wave's 16 A-rows for all K into registers.
    const _Float16* arow = &Ash[(wave * 16 + l) * HHK + half * 16];
    v16h a[4];
#pragma unroll
    for (int kc = 0; kc < 4; ++kc) {
        v8h lo = *(const v8h*)(arow + kc * 32);
        v8h hi = *(const v8h*)(arow + kc * 32 + 8);
        a[kc] = __builtin_shufflevector(lo, hi,
                0, 1, 2, 3, 4, 5, 6, 7, 8, 9, 10, 11, 12, 13, 14, 15);
    }

    v8f acc[8] = {};
#pragma unroll
    for (int nt = 0; nt < 8; ++nt) {
        const _Float16* brow = &Bsh[(nt * 16 + l) * HHK + half * 16];
#pragma unroll
        for (int kc = 0; kc < 4; ++kc) {
            v16h b = *(const v16h*)(brow + kc * 32);
            acc[nt] = __builtin_amdgcn_wmma_f32_16x16x32_f16(
                          false, a[kc], false, b, (short)0, acc[nt], false, false);
        }
    }

    // Streaming (non-temporal) stores: pred is write-once, never re-read.
#pragma unroll
    for (int nt = 0; nt < 8; ++nt) {
        float* cr = C + (size_t)(m_blk + wave * 16 + half * 8) * N
                      + (n_blk + nt * 16 + l);
#pragma unroll
        for (int r = 0; r < 8; ++r)
            __builtin_nontemporal_store(acc[nt][r], cr + (size_t)r * N);
    }
}

// ---------------------------------------------------------------------------
extern "C" void kernel_launch(void* const* d_in, const int* in_sizes, int n_in,
                              void* d_out, int out_size, void* d_ws, size_t ws_size,
                              hipStream_t stream) {
    const float* x   = (const float*)d_in[0];
    const int*   ei  = (const int*)d_in[1];
    const float* W1  = (const float*)d_in[2];
    const float* a1s = (const float*)d_in[3];
    const float* a1d = (const float*)d_in[4];
    const float* b1  = (const float*)d_in[5];
    const float* W2  = (const float*)d_in[6];
    const float* a2s = (const float*)d_in[7];
    const float* a2d = (const float*)d_in[8];
    const float* b2  = (const float*)d_in[9];
    float* out = (float*)d_out;

    const int HID  = in_sizes[3];            // 128
    const int IND  = in_sizes[2] / HID;      // 256
    const int N    = in_sizes[0] / IND;      // 8192
    const int E    = in_sizes[1] / 2;        // 262144
    const int Etot = E + N;                  // self loops appended
    const int* src = ei;
    const int* dst = ei + E;

    // Workspace carve-up (256B aligned slices).
    char* ws = (char*)d_ws;
    auto carve = [&](size_t bytes) { char* p = ws; ws += (bytes + 255) & ~(size_t)255; return p; };
    float*    hbuf0 = (float*)carve((size_t)N * HID * sizeof(float));
    float*    hbuf1 = (float*)carve((size_t)N * HID * sizeof(float));
    _Float16* h16   = (_Float16*)carve((size_t)N * HID * sizeof(_Float16));
    float*    As    = (float*)carve((size_t)N * sizeof(float));
    float*    Ad    = (float*)carve((size_t)N * sizeof(float));
    float*    denom = (float*)carve((size_t)N * sizeof(float));
    unsigned* mkey  = (unsigned*)carve((size_t)N * sizeof(unsigned));
    float*    eV    = (float*)carve((size_t)Etot * sizeof(float));
    float*    exV   = (float*)carve((size_t)Etot * sizeof(float));
    (void)ws_size; (void)n_in; (void)out_size;

    const int B = 256;
    auto blocks = [&](long long total) { return (unsigned)((total + B - 1) / B); };
    const long long NM = (long long)N * HID;
    const long long EC = (long long)Etot * HID;

    // ---- Layer 1 ----
    gemm_f32<<<blocks(NM), B, 0, stream>>>(x, W1, hbuf0, N, IND, HID);
    node_alphas<<<blocks(N), B, 0, stream>>>(hbuf0, a1s, a1d, As, Ad, N, HID);
    init_layer<<<blocks(NM), B, 0, stream>>>(hbuf1, denom, mkey, N, HID);
    edge_logits_max<<<blocks(Etot), B, 0, stream>>>(src, dst, As, Ad, eV, mkey, E, Etot);
    edge_expsum<<<blocks(Etot), B, 0, stream>>>(dst, eV, mkey, exV, denom, E, Etot);
    edge_aggregate<<<blocks(EC), B, 0, stream>>>(src, dst, exV, denom, hbuf0, hbuf1, E, Etot, HID);
    bias_act<<<blocks(NM), B, 0, stream>>>(hbuf1, b1, hbuf0, (_Float16*)nullptr, N, HID);

    // ---- Layer 2 ----
    gemm_f32<<<blocks(NM), B, 0, stream>>>(hbuf0, W2, hbuf1, N, HID, HID);
    node_alphas<<<blocks(N), B, 0, stream>>>(hbuf1, a2s, a2d, As, Ad, N, HID);
    init_layer<<<blocks(NM), B, 0, stream>>>(hbuf0, denom, mkey, N, HID);
    edge_logits_max<<<blocks(Etot), B, 0, stream>>>(src, dst, As, Ad, eV, mkey, E, Etot);
    edge_expsum<<<blocks(Etot), B, 0, stream>>>(dst, eV, mkey, exV, denom, E, Etot);
    edge_aggregate<<<blocks(EC), B, 0, stream>>>(src, dst, exV, denom, hbuf1, hbuf0, E, Etot, HID);
    bias_act<<<blocks(NM), B, 0, stream>>>(hbuf0, b2, (float*)nullptr, h16, N, HID);

    // ---- pred = h @ h^T via LDS-blocked WMMA (read/write balanced @268MB) ----
    dim3 grid(N / 128, N / 128);
    hhT_wmma<<<grid, dim3(256), 0, stream>>>(h16, out, N);
}
